// LocalGCN_48687749267884
// MI455X (gfx1250) — matrix operand order repaired and checked by reading, not compiled, not run on hardware
//
#include <hip/hip_runtime.h>
#include <stdint.h>

#define NN      50000       // nodes
#define NE      800000      // edges (before self loops)
#define ET      850000      // edges + self loops
#define H1N     4           // heads layer 1
#define C1N     64          // channels/head layer 1
#define F1N     256         // H1N*C1N
#define F2N     128         // layer-2 feature dim (heads=1)
#define NGR     100         // graphs
#define NSLOPE  0.2f
#define EPSV    1e-16f
#define NEG_FLT_MAX (-3.402823466e38f)

typedef __attribute__((ext_vector_type(16))) __bf16 v16bf;
typedef __attribute__((ext_vector_type(8)))  float  v8f;

union BFrag { uint4 q[2]; uint32_t u[8]; v16bf v; };

__device__ __forceinline__ void atomic_max_f32(float* addr, float val) {
  if (val >= 0.0f) atomicMax((int*)addr, __float_as_int(val));
  else             atomicMin((unsigned int*)addr, __float_as_uint(val));
}

__global__ __launch_bounds__(256) void fill_f32(float* __restrict__ p, float v, int n) {
  int i = blockIdx.x * 256 + threadIdx.x;
  if (i < n) p[i] = v;
}

// Pack fp32 weights [256 x (NCT*16)] into bf16 hi/lo fragments laid out exactly as
// WMMA B fragments in contiguous K order (same order as the A fragments): dword index ((kt*NCT+ct)*32+lane)*8+v
// holds K = kt*32 + (lane>=16)*16 + 2v .. +1 for column n = ct*16 + lane%16.
template <int NCT>
__global__ __launch_bounds__(256) void pack_w(const float* __restrict__ W,
                                              uint32_t* __restrict__ hi,
                                              uint32_t* __restrict__ lo) {
  int tid = blockIdx.x * 256 + threadIdx.x;
  if (tid >= 8 * NCT * 256) return;
  int v    = tid & 7;
  int lane = (tid >> 3) & 31;
  int ct   = (tid >> 8) % NCT;
  int kt   = tid / (256 * NCT);
  const int NC = NCT * 16;
  int n = ct * 16 + (lane & 15);
  int k = kt * 32 + (lane >> 4) * 16 + 2 * v;
  float w0 = W[k * NC + n];
  float w1 = W[(k + 1) * NC + n];
  __bf16 h0 = (__bf16)w0, h1 = (__bf16)w1;
  __bf16 g0 = (__bf16)(w0 - (float)h0), g1 = (__bf16)(w1 - (float)h1);
  hi[tid] = ((uint32_t)__builtin_bit_cast(unsigned short, h1) << 16) |
            (uint32_t)__builtin_bit_cast(unsigned short, h0);
  lo[tid] = ((uint32_t)__builtin_bit_cast(unsigned short, g1) << 16) |
            (uint32_t)__builtin_bit_cast(unsigned short, g0);
}

// C[N x (NCT*16)] = A[N x 256] * B, with B pre-packed (hi/lo bf16 fragments).
// One wave per 16-row stripe; wave keeps all NCT 16x16 f32 accumulators and
// streams A exactly once. Split-precision: hi*Bhi + lo*Bhi + hi*Blo.
template <int NCT>
__global__ __launch_bounds__(256) void gemm_wmma(const float* __restrict__ A,
                                                 const uint32_t* __restrict__ Bhi,
                                                 const uint32_t* __restrict__ Blo,
                                                 float* __restrict__ C, int nRowTiles) {
  const int ldC = NCT * 16;
  int wave = threadIdx.x >> 5;
  int lane = threadIdx.x & 31;
  int rt = blockIdx.x * 8 + wave;
  if (rt >= nRowTiles) return;         // wave-uniform exit: EXEC all ones for WMMA
  int m    = lane & 15;
  int half = lane >> 4;
  const float* arow = A + (size_t)(rt * 16 + m) * 256;

  v8f z = {};
  v8f acc[NCT];
  #pragma unroll
  for (int i = 0; i < NCT; ++i) acc[i] = z;

  for (int kt = 0; kt < 8; ++kt) {
    // A fragment in contiguous K order (matches packed B): VGPR v -> K = half*16+2v.
    v16bf ah, al;
    #pragma unroll
    for (int v = 0; v < 8; ++v) {
      int koff = half * 16 + 2 * v;
      float2 f = *(const float2*)(arow + kt * 32 + koff);
      __bf16 b0 = (__bf16)f.x, b1 = (__bf16)f.y;
      ah[2 * v]     = b0;
      ah[2 * v + 1] = b1;
      al[2 * v]     = (__bf16)(f.x - (float)b0);
      al[2 * v + 1] = (__bf16)(f.y - (float)b1);
    }
    #pragma unroll
    for (int ct = 0; ct < NCT; ++ct) {
      const uint4* bp = (const uint4*)(Bhi + ((size_t)(kt * NCT + ct) * 32 + lane) * 8);
      const uint4* bq = (const uint4*)(Blo + ((size_t)(kt * NCT + ct) * 32 + lane) * 8);
      BFrag bh, bl;
      bh.q[0] = bp[0]; bh.q[1] = bp[1];
      bl.q[0] = bq[0]; bl.q[1] = bq[1];
      acc[ct] = __builtin_amdgcn_wmma_f32_16x16x32_bf16(false, ah, false, bh.v, (short)0, acc[ct], false, false);
      acc[ct] = __builtin_amdgcn_wmma_f32_16x16x32_bf16(false, al, false, bh.v, (short)0, acc[ct], false, false);
      acc[ct] = __builtin_amdgcn_wmma_f32_16x16x32_bf16(false, ah, false, bl.v, (short)0, acc[ct], false, false);
    }
  }
  // C/D layout: lane holds rows M = half*8 + r at column N = lane%16.
  #pragma unroll
  for (int ct = 0; ct < NCT; ++ct) {
    #pragma unroll
    for (int r = 0; r < 8; ++r) {
      C[(size_t)(rt * 16 + half * 8 + r) * ldC + ct * 16 + m] = acc[ct][r];
    }
  }
}

// a_src[n][h] = sum_c h[n,h,c]*att_src[h,c]; one wave per node, shuffle reduce.
template <int HEADS, int CH>
__global__ __launch_bounds__(256) void att_coef(const float* __restrict__ h,
                                                const float* __restrict__ att_src,
                                                const float* __restrict__ att_dst,
                                                float* __restrict__ a_src,
                                                float* __restrict__ a_dst, int nNodes) {
  const int F = HEADS * CH;
  int wave = threadIdx.x >> 5;
  int lane = threadIdx.x & 31;
  int node = blockIdx.x * 8 + wave;
  if (node >= nNodes) return;
  float ps[HEADS], pd[HEADS];
  #pragma unroll
  for (int i = 0; i < HEADS; ++i) { ps[i] = 0.f; pd[i] = 0.f; }
  #pragma unroll
  for (int j = 0; j < F / 32; ++j) {
    const int hd = (j * 32) / CH;       // constant per unrolled j (32 | CH boundaries)
    int c = j * 32 + lane;
    float hv = h[(size_t)node * F + c];
    ps[hd] += hv * att_src[c];
    pd[hd] += hv * att_dst[c];
  }
  #pragma unroll
  for (int hd = 0; hd < HEADS; ++hd) {
    float s = ps[hd], d = pd[hd];
    #pragma unroll
    for (int off = 16; off > 0; off >>= 1) {
      s += __shfl_xor(s, off, 32);
      d += __shfl_xor(d, off, 32);
    }
    if (lane == 0) {
      a_src[node * HEADS + hd] = s;
      a_dst[node * HEADS + hd] = d;
    }
  }
}

__device__ __forceinline__ void edge_sd(int e, const long long* __restrict__ ei,
                                        int& s, int& d) {
  if (e < NE) { s = (int)ei[e]; d = (int)ei[NE + e]; }
  else        { s = e - NE; d = e - NE; }       // self loops appended
}

template <int H>
__global__ __launch_bounds__(256) void edge_alpha_max(const long long* __restrict__ ei,
                                                      const float* __restrict__ a_src,
                                                      const float* __restrict__ a_dst,
                                                      float* __restrict__ alphaE,
                                                      float* __restrict__ amax) {
  int tid = blockIdx.x * 256 + threadIdx.x;
  if (tid >= ET * H) return;
  int e = tid / H, hd = tid % H;
  int s, d; edge_sd(e, ei, s, d);
  float a = a_src[s * H + hd] + a_dst[d * H + hd];
  a = (a > 0.f) ? a : NSLOPE * a;
  alphaE[tid] = a;
  atomic_max_f32(amax + d * H + hd, a);
}

template <int H>
__global__ __launch_bounds__(256) void edge_exp_denom(const long long* __restrict__ ei,
                                                      float* __restrict__ alphaE,
                                                      const float* __restrict__ amax,
                                                      float* __restrict__ denom) {
  int tid = blockIdx.x * 256 + threadIdx.x;
  if (tid >= ET * H) return;
  int e = tid / H, hd = tid % H;
  int s, d; edge_sd(e, ei, s, d);
  (void)s;
  float w = __expf(alphaE[tid] - amax[d * H + hd]);
  alphaE[tid] = w;
  atomicAdd(denom + d * H + hd, w);
}

template <int H, int CH>
__global__ __launch_bounds__(256) void edge_scatter(const long long* __restrict__ ei,
                                                    const float* __restrict__ h,
                                                    const float* __restrict__ alphaE,
                                                    const float* __restrict__ denom,
                                                    float* __restrict__ out) {
  const int F = H * CH, Q = F / 4;
  int tid = blockIdx.x * 256 + threadIdx.x;
  if (tid >= ET * Q) return;
  int e = tid / Q, q = tid - e * Q;
  int c0 = q * 4, hd = c0 / CH;
  int s, d; edge_sd(e, ei, s, d);
  float w = alphaE[e * H + hd] / (denom[d * H + hd] + EPSV);
  float4 hv = *(const float4*)(h + (size_t)s * F + c0);
  float* o = out + (size_t)d * F + c0;
  atomicAdd(o + 0, hv.x * w);
  atomicAdd(o + 1, hv.y * w);
  atomicAdd(o + 2, hv.z * w);
  atomicAdd(o + 3, hv.w * w);
}

__global__ __launch_bounds__(256) void bias_relu(const float* __restrict__ in,
                                                 const float* __restrict__ bias,
                                                 float* __restrict__ out, int n, int F) {
  int i = blockIdx.x * 256 + threadIdx.x;
  if (i < n) out[i] = fmaxf(in[i] + bias[i % F], 0.f);
}

__global__ __launch_bounds__(256) void pool_nodes(const float* __restrict__ in,
                                                  const float* __restrict__ bias,
                                                  const long long* __restrict__ batch,
                                                  float* __restrict__ gsum,
                                                  float* __restrict__ gcnt) {
  int i = blockIdx.x * 256 + threadIdx.x;
  if (i >= NN * F2N) return;
  int node = i / F2N, c = i - node * F2N;
  float v = fmaxf(in[i] + bias[c], 0.f);
  int g = (int)batch[node];
  atomicAdd(gsum + g * F2N + c, v);
  if (c == 0) atomicAdd(gcnt + g, 1.0f);
}

__global__ __launch_bounds__(256) void finalize_pool(const float* __restrict__ gsum,
                                                     const float* __restrict__ gcnt,
                                                     float* __restrict__ out) {
  int i = blockIdx.x * 256 + threadIdx.x;
  if (i >= NGR * F2N) return;
  out[i] = gsum[i] / fmaxf(gcnt[i / F2N], 1.0f);
}

static inline unsigned int nblk(long long n) { return (unsigned int)((n + 255) / 256); }

extern "C" void kernel_launch(void* const* d_in, const int* in_sizes, int n_in,
                              void* d_out, int out_size, void* d_ws, size_t ws_size,
                              hipStream_t stream) {
  (void)in_sizes; (void)n_in; (void)out_size; (void)ws_size;
  const float*     x        = (const float*)d_in[0];
  const float*     W1       = (const float*)d_in[1];
  const float*     att_src1 = (const float*)d_in[2];
  const float*     att_dst1 = (const float*)d_in[3];
  const float*     bias1    = (const float*)d_in[4];
  const float*     W2       = (const float*)d_in[5];
  const float*     att_src2 = (const float*)d_in[6];
  const float*     att_dst2 = (const float*)d_in[7];
  const float*     bias2    = (const float*)d_in[8];
  const long long* ei       = (const long long*)d_in[9];
  const long long* batch    = (const long long*)d_in[10];

  char* wp = (char*)d_ws;
  auto alloc = [&](size_t bytes) -> void* {
    void* p = (void*)wp;
    wp += (bytes + 255) & ~(size_t)255;
    return p;
  };
  float* h1    = (float*)alloc(sizeof(float) * (size_t)NN * F1N);
  float* out1  = (float*)alloc(sizeof(float) * (size_t)NN * F1N);
  float* h2    = (float*)alloc(sizeof(float) * (size_t)NN * F2N);
  float* out2  = (float*)alloc(sizeof(float) * (size_t)NN * F2N);
  float* as1   = (float*)alloc(sizeof(float) * (size_t)NN * H1N);
  float* ad1   = (float*)alloc(sizeof(float) * (size_t)NN * H1N);
  float* amax1 = (float*)alloc(sizeof(float) * (size_t)NN * H1N);
  float* den1  = (float*)alloc(sizeof(float) * (size_t)NN * H1N);
  float* as2   = (float*)alloc(sizeof(float) * (size_t)NN);
  float* ad2   = (float*)alloc(sizeof(float) * (size_t)NN);
  float* amax2 = (float*)alloc(sizeof(float) * (size_t)NN);
  float* den2  = (float*)alloc(sizeof(float) * (size_t)NN);
  float* aE1   = (float*)alloc(sizeof(float) * (size_t)ET * H1N);
  float* aE2   = (float*)alloc(sizeof(float) * (size_t)ET);
  uint32_t* w1hi = (uint32_t*)alloc(sizeof(uint32_t) * 8 * 16 * 256);
  uint32_t* w1lo = (uint32_t*)alloc(sizeof(uint32_t) * 8 * 16 * 256);
  uint32_t* w2hi = (uint32_t*)alloc(sizeof(uint32_t) * 8 * 8 * 256);
  uint32_t* w2lo = (uint32_t*)alloc(sizeof(uint32_t) * 8 * 8 * 256);
  float* gsum  = (float*)alloc(sizeof(float) * NGR * F2N);
  float* gcnt  = (float*)alloc(sizeof(float) * NGR);

  // ---- init accumulated buffers (ws is poisoned; must init every call)
  fill_f32<<<nblk((long long)NN * F1N), 256, 0, stream>>>(out1, 0.f, NN * F1N);
  fill_f32<<<nblk((long long)NN * F2N), 256, 0, stream>>>(out2, 0.f, NN * F2N);
  fill_f32<<<nblk(NN * H1N), 256, 0, stream>>>(den1, 0.f, NN * H1N);
  fill_f32<<<nblk(NN), 256, 0, stream>>>(den2, 0.f, NN);
  fill_f32<<<nblk(NN * H1N), 256, 0, stream>>>(amax1, NEG_FLT_MAX, NN * H1N);
  fill_f32<<<nblk(NN), 256, 0, stream>>>(amax2, NEG_FLT_MAX, NN);
  fill_f32<<<nblk(NGR * F2N), 256, 0, stream>>>(gsum, 0.f, NGR * F2N);
  fill_f32<<<nblk(NGR), 256, 0, stream>>>(gcnt, 0.f, NGR);

  // ---- pack weights into WMMA B-fragment layout (bf16 hi/lo)
  pack_w<16><<<nblk(8 * 16 * 256), 256, 0, stream>>>(W1, w1hi, w1lo);
  pack_w<8><<<nblk(8 * 8 * 256), 256, 0, stream>>>(W2, w2hi, w2lo);

  const int rowTiles = NN / 16;   // 3125

  // ---- layer 1
  gemm_wmma<16><<<nblk((long long)rowTiles * 32), 256, 0, stream>>>(x, w1hi, w1lo, h1, rowTiles);
  att_coef<H1N, C1N><<<nblk((long long)NN * 32), 256, 0, stream>>>(h1, att_src1, att_dst1, as1, ad1, NN);
  edge_alpha_max<H1N><<<nblk((long long)ET * H1N), 256, 0, stream>>>(ei, as1, ad1, aE1, amax1);
  edge_exp_denom<H1N><<<nblk((long long)ET * H1N), 256, 0, stream>>>(ei, aE1, amax1, den1);
  edge_scatter<H1N, C1N><<<nblk((long long)ET * (F1N / 4)), 256, 0, stream>>>(ei, h1, aE1, den1, out1);
  bias_relu<<<nblk((long long)NN * F1N), 256, 0, stream>>>(out1, bias1, h1, NN * F1N, F1N);

  // ---- layer 2 (input = relu'd aggregate, now stored in h1)
  gemm_wmma<8><<<nblk((long long)rowTiles * 32), 256, 0, stream>>>(h1, w2hi, w2lo, h2, rowTiles);
  att_coef<1, F2N><<<nblk((long long)NN * 32), 256, 0, stream>>>(h2, att_src2, att_dst2, as2, ad2, NN);
  edge_alpha_max<1><<<nblk(ET), 256, 0, stream>>>(ei, as2, ad2, aE2, amax2);
  edge_exp_denom<1><<<nblk(ET), 256, 0, stream>>>(ei, aE2, amax2, den2);
  edge_scatter<1, F2N><<<nblk((long long)ET * (F2N / 4)), 256, 0, stream>>>(ei, h2, aE2, den2, out2);

  // ---- bias + relu + global mean pool
  pool_nodes<<<nblk((long long)NN * F2N), 256, 0, stream>>>(out2, bias2, batch, gsum, gcnt);
  finalize_pool<<<nblk(NGR * F2N), 256, 0, stream>>>(gsum, gcnt, (float*)d_out);
}
